// Net_53910429499748
// MI455X (gfx1250) — matrix operand order repaired and checked by reading, not compiled, not run on hardware
//
#include <hip/hip_runtime.h>
#include <math.h>

typedef __attribute__((ext_vector_type(16))) _Float16 v16h;
typedef __attribute__((ext_vector_type(8)))  _Float16 v8h;
typedef __attribute__((ext_vector_type(8)))  float    v8f;

#define BATCH   4096
#define HIDDEN  500

__device__ __forceinline__ float trunc13(float v) {
    return __int_as_float(__float_as_int(v) & 0xFFFFE000);
}

__device__ __forceinline__ v8f wmma_f16(v16h a, v16h b, v8f c) {
    return __builtin_amdgcn_wmma_f32_16x16x32_f16(false, a, false, b, (short)0, c,
                                                  false, false);
}

// Assemble A fragment (16x32 f16) from two aligned 16B loads.
// Lane L: row = L&15, kbase = (L&16 ? 8 : 0); halves 0..7 -> K=kbase..kbase+7,
// halves 8..15 -> K=kbase+16..kbase+23 (ISA 7.12.2, 16-bit A 16x32 layout).
__device__ __forceinline__ v16h a_frag_from(const _Float16* p) {
    v8h lo = *(const v8h*)p;
    v8h hi = *(const v8h*)(p + 16);
    return __builtin_shufflevector(lo, hi, 0,1,2,3,4,5,6,7,8,9,10,11,12,13,14,15);
}

// ---------------------------------------------------------------------------
// Weight packing into B-fragment layout:
// B operand (32x16 f16), lane L: col = L&15, K = (L&16?8:0)+(i&7)+(i&8?16:0) for half i=0..15.
// pack index = ((kt*NT + nt)*32 + lane)*16 + i
// ---------------------------------------------------------------------------
__global__ void pack_w_conv1(const float* __restrict__ w, _Float16* __restrict__ pack) {
    int idx = blockIdx.x * 256 + threadIdx.x;              // 2*32*16 = 1024
    if (idx >= 2 * 32 * 16) return;
    int i = idx & 15, lane = (idx >> 4) & 31, nt = idx >> 9;
    int n = nt * 16 + (lane & 15);
    int k = ((lane & 16) ? 8 : 0) + (i & 7) + ((i & 8) ? 16 : 0); // k = dy*5+dx
    float v = (n < 20 && k < 25) ? w[n * 25 + k] : 0.f;    // [20][1][5][5]
    pack[idx] = (_Float16)v;
}

__global__ void pack_w_conv2(const float* __restrict__ w, _Float16* __restrict__ pack) {
    int idx = blockIdx.x * 256 + threadIdx.x;              // 16*4*32*16 = 32768
    if (idx >= 16 * 4 * 32 * 16) return;
    int i = idx & 15, lane = (idx >> 4) & 31, nt = (idx >> 9) & 3, kt = idx >> 11;
    int n = nt * 16 + (lane & 15);
    int k = kt * 32 + ((lane & 16) ? 8 : 0) + (i & 7) + ((i & 8) ? 16 : 0);
    float v = 0.f;
    if (n < 50 && k < 500) {
        int c = k / 25, r = k - 25 * c, dy = r / 5, dx = r - 5 * dy;
        v = w[((n * 20 + c) * 5 + dy) * 5 + dx];           // OIHW [50][20][5][5]
    }
    pack[idx] = (_Float16)v;
}

__global__ void pack_w_fc1(const float* __restrict__ w, _Float16* __restrict__ pack) {
    int idx = blockIdx.x * 256 + threadIdx.x;              // 25*32*32*16 = 409600
    if (idx >= 25 * 32 * 32 * 16) return;
    int i = idx & 15, lane = (idx >> 4) & 31, nt = (idx >> 9) & 31, kt = idx >> 14;
    int n = nt * 16 + (lane & 15);
    int k = kt * 32 + ((lane & 16) ? 8 : 0) + (i & 7) + ((i & 8) ? 16 : 0); // k < 800 always
    float v = (n < HIDDEN) ? w[n * 800 + k] : 0.f;         // B[k][n] = w_fc1[n][k]
    pack[idx] = (_Float16)v;
}

__global__ void pack_w_fc2(const float* __restrict__ w, _Float16* __restrict__ pack) {
    int idx = blockIdx.x * 256 + threadIdx.x;              // 16*1*32*16 = 8192
    if (idx >= 16 * 32 * 16) return;
    int i = idx & 15, lane = (idx >> 4) & 31, kt = idx >> 9;
    int n = lane & 15;
    int k = kt * 32 + ((lane & 16) ? 8 : 0) + (i & 7) + ((i & 8) ? 16 : 0);
    float v = (n < 10 && k < HIDDEN) ? w[n * HIDDEN + k] : 0.f;
    pack[idx] = (_Float16)v;
}

// ---------------------------------------------------------------------------
// conv1 per image as WMMA: im2col in LDS (576x32, K=25 pad 32, single ktile),
// 36 mtiles x 2 ntiles = 72 one-WMMA tiles, trunc13+relu -> LDS -> 2x2 pool.
// ntile = wave&1 is wave-constant, so B fragment and bias are hoisted out of
// the tile loop; body is ds_load_b128 x2 -> wmma -> ds stores.
// pooled layout [b][c][12][12] f16.
// ---------------------------------------------------------------------------
__global__ void __launch_bounds__(512) conv1_pool_wmma(const float* __restrict__ x,
                                                       const _Float16* __restrict__ pack,
                                                       const float* __restrict__ bias,
                                                       _Float16* __restrict__ pooled) {
    __shared__ __attribute__((aligned(16))) _Float16 simg[784];       // 28x28
    __shared__ __attribute__((aligned(16))) _Float16 A2[576 * 32];    // 36 KB
    __shared__ __attribute__((aligned(16))) _Float16 sC[576 * 32];    // 36 KB
    int b = blockIdx.x;
    const float* src = x + (size_t)b * 784;
    for (int t = threadIdx.x; t < 784; t += 512) simg[t] = (_Float16)src[t];
    __syncthreads();

    // Build im2col rows: A2[m][k] = simg[(oy+dy)*28 + ox+dx], k = dy*5+dx.
    // Offsets are compile-time constants after unrolling -> ds_load immediates.
    for (int m = threadIdx.x; m < 576; m += 512) {
        int oy = m / 24, ox = m - 24 * oy;
        int base = oy * 28 + ox;
        v8h h[4];
#pragma unroll
        for (int k = 0; k < 32; k++) {
            _Float16 v = (_Float16)0.f;
            if (k < 25) v = simg[base + (k / 5) * 28 + (k - 5 * (k / 5))];
            h[k >> 3][k & 7] = v;
        }
        v8h* dst = (v8h*)(A2 + m * 32);
#pragma unroll
        for (int q = 0; q < 4; q++) dst[q] = h[q];
    }
    __syncthreads();

    int wave = threadIdx.x >> 5, lane = threadIdx.x & 31;
    int kb8 = (lane & 16) ? 8 : 0, rbase = (lane & 16) ? 8 : 0;
    int ntile = wave & 1;                                  // loop-invariant per wave
    v16h bf = *(const v16h*)(pack + (size_t)(ntile * 32 + lane) * 16);
    int n = ntile * 16 + (lane & 15);
    float bn = (n < 20) ? bias[n] : 0.f;

    for (int mtile = wave >> 1; mtile < 36; mtile += 8) {
        int mrow = mtile * 16 + (lane & 15);
        v16h a = a_frag_from(A2 + mrow * 32 + kb8);
        v8f acc = {};
        acc = wmma_f16(a, bf, acc);
#pragma unroll
        for (int r = 0; r < 8; r++) {
            float v = trunc13(acc[r] + bn);
            v = v > 0.f ? v : 0.f;
            sC[(mtile * 16 + rbase + r) * 32 + n] = (_Float16)v;
        }
    }
    __syncthreads();

    // 2x2 maxpool 24x24 -> 12x12, channels 0..19
    for (int t = threadIdx.x; t < 2880; t += 512) {
        int px = t % 12, py = (t / 12) % 12, c = t / 144;
        int m0 = (2 * py) * 24 + 2 * px;
        float v = fmaxf(fmaxf((float)sC[m0 * 32 + c], (float)sC[(m0 + 1) * 32 + c]),
                        fmaxf((float)sC[(m0 + 24) * 32 + c], (float)sC[(m0 + 25) * 32 + c]));
        pooled[(size_t)b * 2880 + t] = (_Float16)v;
    }
}

// ---------------------------------------------------------------------------
// conv2 fused per image: im2col materialized once in LDS (64x512 f16, 64 KB),
// each thread owns one K-column so the (c,dy,dx) decode happens once/thread.
// 16 waves x (16x16) WMMA tiles; k-loop software-pipelined by one stage so
// fragment loads for kt+1 are in flight across the kt WMMA.
// bias + trunc13 + relu + 2x2 pool -> flat [b][800] f16.
// ---------------------------------------------------------------------------
__global__ void __launch_bounds__(512) conv2_pool(const _Float16* __restrict__ pooled1,
                                                  const _Float16* __restrict__ pack,
                                                  const float* __restrict__ bias,
                                                  _Float16* __restrict__ A3) {
    __shared__ __attribute__((aligned(16))) _Float16 simg[2880];      // [20][12][12]
    __shared__ __attribute__((aligned(16))) _Float16 A2[64 * 512];    // 64 KB
    __shared__ __attribute__((aligned(16))) float    sC[64 * 64];     // 16 KB
    int b = blockIdx.x;
    for (int t = threadIdx.x; t < 2880; t += 512) simg[t] = pooled1[(size_t)b * 2880 + t];
    __syncthreads();

    // Thread t owns column k=t: A2[m][k] = simg[c*144 + (y+dy)*12 + (x+dx)],
    // m = y*8+x, k = c*25 + dy*5 + dx (k >= 500 -> zero padding).
    {
        int k = threadIdx.x;
        int koff = -1;
        if (k < 500) {
            int c = k / 25, r = k - 25 * c, dy = r / 5, dx = r - 5 * dy;
            koff = c * 144 + dy * 12 + dx;
        }
#pragma unroll 8
        for (int m = 0; m < 64; m++) {
            int yx = (m >> 3) * 12 + (m & 7);
            A2[m * 512 + k] = (koff >= 0) ? simg[koff + yx] : (_Float16)0.f;
        }
    }
    __syncthreads();

    int wave = threadIdx.x >> 5, lane = threadIdx.x & 31;
    int mtile = wave >> 2, ntile = wave & 3;
    int mrow = mtile * 16 + (lane & 15);
    int kb8 = (lane & 16) ? 8 : 0;
    const _Float16* arow = A2 + mrow * 512 + kb8;

    v16h a  = a_frag_from(arow);
    v16h bf = *(const v16h*)(pack + (size_t)(ntile * 32 + lane) * 16);
    v8f acc = {};
    for (int kt = 0; kt < 15; kt++) {
        v16h an  = a_frag_from(arow + (kt + 1) * 32);
        v16h bfn = *(const v16h*)(pack + (size_t)(((kt + 1) * 4 + ntile) * 32 + lane) * 16);
        acc = wmma_f16(a, bf, acc);
        a = an; bf = bfn;
    }
    acc = wmma_f16(a, bf, acc);

    int n = ntile * 16 + (lane & 15);
    float bn = (n < 50) ? bias[n] : 0.f;
    int rbase = (lane & 16) ? 8 : 0;
#pragma unroll
    for (int r = 0; r < 8; r++) {
        int m = mtile * 16 + rbase + r;
        float v = trunc13(acc[r] + bn);
        v = v > 0.f ? v : 0.f;
        sC[m * 64 + n] = v;
    }
    __syncthreads();

    // 2x2 maxpool over 8x8 -> [50][4][4], flattened c*16 + py*4 + px
    for (int t = threadIdx.x; t < 800; t += 512) {
        int c = t >> 4, q = t & 15, py = q >> 2, px = q & 3;
        int m0 = (2 * py) * 8 + 2 * px;
        float v = fmaxf(fmaxf(sC[m0 * 64 + c], sC[(m0 + 1) * 64 + c]),
                        fmaxf(sC[(m0 + 8) * 64 + c], sC[(m0 + 9) * 64 + c]));
        A3[(size_t)b * 800 + t] = (_Float16)v;
    }
}

// ---------------------------------------------------------------------------
// fc1: [4096x800] @ [800x512(pad)] WMMA GEMM, software-pipelined k-loop,
// bias + trunc13 + relu, f16 out.
// grid (256 mtiles, 8), block 128 = 4 waves; wave ntile = blockIdx.y*4+wave.
// ---------------------------------------------------------------------------
__global__ void __launch_bounds__(128) fc1_gemm(const _Float16* __restrict__ A,
                                                const _Float16* __restrict__ pack,
                                                const float* __restrict__ bias,
                                                _Float16* __restrict__ out) {
    int wave = threadIdx.x >> 5, lane = threadIdx.x & 31;
    int mtile = blockIdx.x;
    int ntile = blockIdx.y * 4 + wave;
    int m = mtile * 16 + (lane & 15);
    int kb8 = (lane & 16) ? 8 : 0;
    const _Float16* arow = A + (size_t)m * 800 + kb8;

    v16h a  = a_frag_from(arow);
    v16h bf = *(const v16h*)(pack + (size_t)(ntile * 32 + lane) * 16);
    v8f acc = {};
    for (int kt = 0; kt < 24; kt++) {
        v16h an  = a_frag_from(arow + (kt + 1) * 32);
        v16h bfn = *(const v16h*)(pack + (size_t)(((kt + 1) * 32 + ntile) * 32 + lane) * 16);
        acc = wmma_f16(a, bf, acc);
        a = an; bf = bfn;
    }
    acc = wmma_f16(a, bf, acc);

    int n = ntile * 16 + (lane & 15);
    float bn = (n < HIDDEN) ? bias[n] : 0.f;
    int rbase = (lane & 16) ? 8 : 0;
#pragma unroll
    for (int r = 0; r < 8; r++) {
        int mm = mtile * 16 + rbase + r;
        float v = trunc13(acc[r] + bn);
        v = v > 0.f ? v : 0.f;
        out[(size_t)mm * 512 + n] = (_Float16)((n < HIDDEN) ? v : 0.f);
    }
}

// ---------------------------------------------------------------------------
// fc2: [4096x512] @ [512x16(pad)] WMMA GEMM (pipelined) + bias + trunc13 +
// log_softmax(10).  block 128 = 4 waves = 4 mtiles = 64 rows; fp32 out.
// ---------------------------------------------------------------------------
__global__ void __launch_bounds__(128) fc2_logsoftmax(const _Float16* __restrict__ A,
                                                      const _Float16* __restrict__ pack,
                                                      const float* __restrict__ bias,
                                                      float* __restrict__ out) {
    __shared__ float sH[64 * 16];
    int wave = threadIdx.x >> 5, lane = threadIdx.x & 31;
    int mtile = blockIdx.x * 4 + wave;
    int m = mtile * 16 + (lane & 15);
    int kb8 = (lane & 16) ? 8 : 0;
    const _Float16* arow = A + (size_t)m * 512 + kb8;

    v16h a  = a_frag_from(arow);
    v16h bf = *(const v16h*)(pack + (size_t)(lane * 16));
    v8f acc = {};
    for (int kt = 0; kt < 15; kt++) {
        v16h an  = a_frag_from(arow + (kt + 1) * 32);
        v16h bfn = *(const v16h*)(pack + (size_t)(((kt + 1) * 32 + lane) * 16));
        acc = wmma_f16(a, bf, acc);
        a = an; bf = bfn;
    }
    acc = wmma_f16(a, bf, acc);

    int n = lane & 15;
    float bn = (n < 10) ? bias[n] : 0.f;
    int rbase = (lane & 16) ? 8 : 0;
#pragma unroll
    for (int r = 0; r < 8; r++)
        sH[(wave * 16 + rbase + r) * 16 + n] = trunc13(acc[r] + bn);
    __syncthreads();

    int t = threadIdx.x;
    if (t < 64) {
        const float* row = &sH[t * 16];
        float mx = row[0];
#pragma unroll
        for (int j = 1; j < 10; j++) mx = fmaxf(mx, row[j]);
        float s = 0.f;
#pragma unroll
        for (int j = 0; j < 10; j++) s += expf(row[j] - mx);
        float lse = mx + logf(s);
        int mm = blockIdx.x * 64 + t;
#pragma unroll
        for (int j = 0; j < 10; j++) out[(size_t)mm * 10 + j] = row[j] - lse;
    }
}

// ---------------------------------------------------------------------------
extern "C" void kernel_launch(void* const* d_in, const int* in_sizes, int n_in,
                              void* d_out, int out_size, void* d_ws, size_t ws_size,
                              hipStream_t stream) {
    const float* x       = (const float*)d_in[0];
    const float* w_conv1 = (const float*)d_in[1];
    const float* b_conv1 = (const float*)d_in[2];
    const float* w_conv2 = (const float*)d_in[3];
    const float* b_conv2 = (const float*)d_in[4];
    const float* w_fc1   = (const float*)d_in[5];
    const float* b_fc1   = (const float*)d_in[6];
    const float* w_fc2   = (const float*)d_in[7];
    const float* b_fc2   = (const float*)d_in[8];
    float* out = (float*)d_out;

    char* ws = (char*)d_ws;
    _Float16* pooled1 = (_Float16*)(ws);                    // 4096*2880 f16 = 23,592,960 B
    _Float16* A3      = (_Float16*)(ws + 23592960);         // 4096*800  f16 =  6,553,600 B
    _Float16* fc1out  = (_Float16*)(ws + 30146560);         // 4096*512  f16 =  4,194,304 B
    _Float16* packc2  = (_Float16*)(ws + 34340864);         // 32768 f16     =     65,536 B
    _Float16* packf1  = (_Float16*)(ws + 34406400);         // 409600 f16    =    819,200 B
    _Float16* packf2  = (_Float16*)(ws + 35225600);         // 8192 f16      =     16,384 B
    _Float16* packc1  = (_Float16*)(ws + 35241984);         // 1024 f16      =      2,048 B
    (void)ws_size; (void)in_sizes; (void)n_in; (void)out_size;

    pack_w_conv1<<<4,    256, 0, stream>>>(w_conv1, packc1);
    pack_w_conv2<<<128,  256, 0, stream>>>(w_conv2, packc2);
    pack_w_fc1  <<<1600, 256, 0, stream>>>(w_fc1,   packf1);
    pack_w_fc2  <<<32,   256, 0, stream>>>(w_fc2,   packf2);

    conv1_pool_wmma<<<BATCH, 512, 0, stream>>>(x, packc1, b_conv1, pooled1);
    conv2_pool<<<BATCH, 512, 0, stream>>>(pooled1, packc2, b_conv2, A3);
    fc1_gemm<<<dim3(256, 8), 128, 0, stream>>>(A3, packf1, b_fc1, fc1out);
    fc2_logsoftmax<<<64, 128, 0, stream>>>(fc1out, packf2, b_fc2, out);
}
